// GraphLabelPropagation_56246891708646
// MI455X (gfx1250) — matrix operand order repaired and checked by reading, not compiled, not run on hardware
//
#include <hip/hip_runtime.h>
#include <hip/hip_bf16.h>
#include <math.h>

// Problem constants (fixed by the reference)
#define N_ROWS   16384
#define DIMX     128
#define KSEL     51        // K+1 kept per row during selection
#define KNN      50        // K edges per row after dropping self
#define NCLS     100
#define ALPHA_C  0.99f
#define TOL_C    1e-6f
#define THRESH_C 0.4f

typedef __attribute__((ext_vector_type(16))) _Float16 v16h;
typedef __attribute__((ext_vector_type(8)))  _Float16 v8h;
typedef __attribute__((ext_vector_type(8)))  float    v8f;
typedef __attribute__((ext_vector_type(4)))  unsigned int u32x4;
typedef __attribute__((ext_vector_type(8)))  int i32x8;
typedef __attribute__((ext_vector_type(4)))  int i32x4;

#if defined(__has_builtin)
#if __has_builtin(__builtin_amdgcn_tensor_load_to_lds) && __has_builtin(__builtin_amdgcn_s_wait_tensorcnt)
#define HAVE_TDM 1
#endif
#endif
#ifndef HAVE_TDM
#define HAVE_TDM 0
#endif

// ---------------------------------------------------------------------------
// 1) Row L2-normalize X (f32) -> Xn (f16).  One wave32 per row.
// ---------------------------------------------------------------------------
__global__ __launch_bounds__(256) void normalizeK(const float* __restrict__ X,
                                                  _Float16* __restrict__ Xn) {
  int gwave = (int)((blockIdx.x * blockDim.x + threadIdx.x) >> 5);
  int lane  = (int)(threadIdx.x & 31);
  if (gwave >= N_ROWS) return;
  const float* xr = X + (size_t)gwave * DIMX;
  float s = 0.f;
  for (int j = lane; j < DIMX; j += 32) { float v = xr[j]; s += v * v; }
  for (int off = 16; off > 0; off >>= 1) s += __shfl_xor(s, off, 32);
  float inv = 1.0f / fmaxf(sqrtf(s), 1e-12f);
  for (int j = lane; j < DIMX; j += 32)
    Xn[(size_t)gwave * DIMX + j] = (_Float16)(xr[j] * inv);
}

#if HAVE_TDM
// Issue one TDM DMA of a contiguous 16x128 f16 tile (4 KB) into LDS.
// D# per CDNA5 ISA ch.8: group0 = {flags/count, lds_addr, global_addr, type=2},
// group1 = {data_size=2B, tensor/tile dims for a 1-D tile of 2048 elements}.
__device__ __forceinline__ void tdm_load_tile(const _Float16* gsrc, _Float16* ldst) {
  const unsigned int E = 16u * DIMX;   // 2048 elements of 2 bytes
  unsigned int lds_addr = (unsigned int)(unsigned long long)ldst; // low 32 bits = LDS offset
  unsigned long long ga = (unsigned long long)gsrc;

  unsigned int g0w0 = 1u;                                          // count=1, user mode
  unsigned int g0w1 = lds_addr;                                    // lds_addr
  unsigned int g0w2 = (unsigned int)(ga & 0xFFFFFFFFull);          // global_addr lo
  unsigned int g0w3 = (unsigned int)((ga >> 32) & 0x01FFFFFFull)   // global_addr hi (57b)
                    | (2u << 30);                                  // type = 2 ("image")

  unsigned int g1w0 = (1u << 16);                 // workgroup_mask=0, data_size=1 (2 bytes)
  unsigned int g1w1 = (E & 0xFFFFu) << 16;        // tensor_dim0 lo16 (bits 63:48)
  unsigned int g1w2 = (E >> 16) | (1u << 16);     // tensor_dim0 hi16 | tensor_dim1 lo16 = 1
  unsigned int g1w3 = (E & 0xFFFFu) << 16;        // tensor_dim1 hi16=0 | tile_dim0 = 2048
  unsigned int g1w4 = 1u;                         // tile_dim1 = 1, tile_dim2 = 0
  unsigned int g1w5 = E;                          // tensor_dim0_stride lo32
  unsigned int g1w6 = 0u;                         // stride hi | tensor_dim1_stride lo
  unsigned int g1w7 = 0u;                         // tensor_dim1_stride hi

  u32x4 g0;
  g0[0] = (unsigned int)__builtin_amdgcn_readfirstlane((int)g0w0);
  g0[1] = (unsigned int)__builtin_amdgcn_readfirstlane((int)g0w1);
  g0[2] = (unsigned int)__builtin_amdgcn_readfirstlane((int)g0w2);
  g0[3] = (unsigned int)__builtin_amdgcn_readfirstlane((int)g0w3);
  i32x8 g1;
  g1[0] = __builtin_amdgcn_readfirstlane((int)g1w0);
  g1[1] = __builtin_amdgcn_readfirstlane((int)g1w1);
  g1[2] = __builtin_amdgcn_readfirstlane((int)g1w2);
  g1[3] = __builtin_amdgcn_readfirstlane((int)g1w3);
  g1[4] = __builtin_amdgcn_readfirstlane((int)g1w4);
  g1[5] = __builtin_amdgcn_readfirstlane((int)g1w5);
  g1[6] = __builtin_amdgcn_readfirstlane((int)g1w6);
  g1[7] = __builtin_amdgcn_readfirstlane((int)g1w7);
  i32x4 z4 = (i32x4)0;
#if __clang_major__ >= 23
  i32x8 z8 = (i32x8)0;
  __builtin_amdgcn_tensor_load_to_lds(g0, g1, z4, z4, z8, 0);
#else
  __builtin_amdgcn_tensor_load_to_lds(g0, g1, z4, z4, 0);
#endif
}
#endif // HAVE_TDM

// ---------------------------------------------------------------------------
// 2) Fused WMMA GEMM (sims = Xn @ Xn^T) + streaming per-row top-51 selection.
//    4 waves/block, each wave owns a 16-row block.  Column tiles are streamed
//    into double-buffered LDS by the Tensor Data Mover (wave 0 issues the DMA
//    for tile t+1 while all waves compute tile t), falling back to a manual
//    cooperative copy when the TDM builtin is unavailable.
// ---------------------------------------------------------------------------
#define GT_WAVES 4
#define NTILES   (N_ROWS / 16)
__global__ __launch_bounds__(128) void gemm_topk(const _Float16* __restrict__ Xn,
                                                 float* __restrict__ topV,
                                                 int*   __restrict__ topI) {
  __shared__ alignas(16) _Float16 tileB[2][16 * DIMX];        // 2 x 4 KB column tiles
  __shared__ float simBuf[GT_WAVES][16][16];                  // per-wave 16x16 sims
  __shared__ float tV[GT_WAVES][16][KSEL];
  __shared__ int   tIx[GT_WAVES][16][KSEL];

  const int tid  = (int)threadIdx.x;
  const int wave = tid >> 5;
  const int lane = tid & 31;
  const int rowBase = ((int)blockIdx.x * GT_WAVES + wave) * 16;
  const int coln = lane & 15;
  const int khalf = (lane >> 4);          // which K half this lane holds

  // init per-row top lists
  if (lane < 16)
    for (int j = 0; j < KSEL; ++j) { tV[wave][lane][j] = -1e30f; tIx[wave][lane][j] = 0; }
  float minVal = -1e30f; int minPos = 0;

  // A fragments for this wave's 16 rows, K = 0..127 (4 fragments of 16x32).
  // 16-bit A 16x32 layout (ISA 7.12.2): lanes 0-15 hold K {0..7,16..23},
  // lanes 16-31 hold K {8..15,24..31}.
  const int rowA = rowBase + coln;
  v16h afrag[4];
#pragma unroll
  for (int kk = 0; kk < 4; ++kk) {
    const _Float16* pa = Xn + (size_t)rowA * DIMX + kk * 32 + khalf * 8;
    v8h lo = *(const v8h*)(pa);
    v8h hi = *(const v8h*)(pa + 16);
    afrag[kk] = __builtin_shufflevector(lo, hi, 0,1,2,3,4,5,6,7,8,9,10,11,12,13,14,15);
  }

#if HAVE_TDM
  if (wave == 0) {
    tdm_load_tile(Xn, &tileB[0][0]);
    __builtin_amdgcn_s_wait_tensorcnt(0);
  }
  __syncthreads();
#endif

  for (int tile = 0; tile < NTILES; ++tile) {
#if HAVE_TDM
    const int cur = tile & 1;
    const _Float16* tb = &tileB[cur][0];
    if (wave == 0 && tile + 1 < NTILES)
      tdm_load_tile(Xn + (size_t)(tile + 1) * 16 * DIMX, &tileB[cur ^ 1][0]);
#else
    const _Float16* tb = &tileB[0][0];
    __syncthreads();
    {
      const uint4* src = (const uint4*)(Xn + (size_t)tile * 16 * DIMX);
      for (int c = tid; c < 256; c += 128) ((uint4*)&tileB[0][0])[c] = src[c];
    }
    __syncthreads();
    if (tile + 1 < NTILES)
      __builtin_prefetch(Xn + (size_t)(tile + 1) * 16 * DIMX, 0, 0);
#endif

    // Load all 4 B fragments first (one DS wait), then issue WMMAs back-to-back.
    // B 32x16 (16-bit): lanes 0-15 hold K {0..7,16..23}, lanes 16-31 hold K {8..15,24..31}.
    v16h bfr[4];
#pragma unroll
    for (int kk = 0; kk < 4; ++kk) {
      const _Float16* pb = tb + coln * DIMX + kk * 32 + khalf * 8;
      v8h b0 = *(const v8h*)(pb);
      v8h b1 = *(const v8h*)(pb + 16);
      bfr[kk] = __builtin_shufflevector(b0, b1, 0,1,2,3,4,5,6,7,8,9,10,11,12,13,14,15);
    }
    v8f acc = {};
#pragma unroll
    for (int kk = 0; kk < 4; ++kk)
      acc = __builtin_amdgcn_wmma_f32_16x16x32_f16(false, afrag[kk], false, bfr[kk],
                                                   (short)0, acc, false, false);

    // stage C tile: lane holds N=coln, VGPR r -> M = r (+8 for lanes 16-31)
#pragma unroll
    for (int i = 0; i < 8; ++i) {
      int m = (lane < 16) ? i : (i + 8);
      simBuf[wave][m][coln] = acc[i];
    }
    __builtin_amdgcn_wave_barrier();

    // per-row top-51 maintenance (lanes 0..15, one row each)
    if (lane < 16) {
      float* rv = tV[wave][lane];
      int*   ri = tIx[wave][lane];
      for (int j = 0; j < 16; ++j) {
        float v = simBuf[wave][lane][j];
        if (v > minVal) {
          rv[minPos] = v; ri[minPos] = tile * 16 + j;
          minVal = rv[0]; minPos = 0;
          for (int q = 1; q < KSEL; ++q) {
            float t = rv[q];
            if (t < minVal) { minVal = t; minPos = q; }
          }
        }
      }
    }
    __builtin_amdgcn_wave_barrier();

#if HAVE_TDM
    __syncthreads();   // everyone done with tileB[cur] / previous buffer
    if (wave == 0 && tile + 1 < NTILES)
      __builtin_amdgcn_s_wait_tensorcnt(0);
    __syncthreads();   // tileB[cur^1] now valid for all waves
#endif
  }

  if (lane < 16) {
    int row = rowBase + lane;
    for (int j = 0; j < KSEL; ++j) {
      topV[(size_t)row * KSEL + j] = tV[wave][lane][j];
      topI[(size_t)row * KSEL + j] = tIx[wave][lane][j];
    }
  }
}

// ---------------------------------------------------------------------------
// 3) Sort each row's 51 candidates desc, drop self (pos 0), emit edges w=v^3,
//    accumulate symmetric degree S with atomics.
// ---------------------------------------------------------------------------
__global__ void sortEdgesK(const float* __restrict__ topV, const int* __restrict__ topI,
                           float* __restrict__ eW, int* __restrict__ eC,
                           float* __restrict__ S) {
  int row = (int)(blockIdx.x * blockDim.x + threadIdx.x);
  if (row >= N_ROWS) return;
  float v[KSEL]; int ix[KSEL];
  for (int j = 0; j < KSEL; ++j) { v[j] = topV[(size_t)row * KSEL + j]; ix[j] = topI[(size_t)row * KSEL + j]; }
  for (int a = 1; a < KSEL; ++a) {           // insertion sort, descending
    float kv = v[a]; int ki = ix[a]; int b = a - 1;
    while (b >= 0 && v[b] < kv) { v[b+1] = v[b]; ix[b+1] = ix[b]; --b; }
    v[b+1] = kv; ix[b+1] = ki;
  }
  for (int t = 0; t < KNN; ++t) {            // entries 1..50 (drop max/self)
    float s = v[t + 1]; int c = ix[t + 1];
    float w = s * s * s;
    if (c == row) w = 0.f;
    eW[(size_t)row * KNN + t] = w;
    eC[(size_t)row * KNN + t] = c;
    if (w != 0.f) { atomicAdd(&S[row], w); atomicAdd(&S[c], w); }
  }
}

__global__ void dinvK(const float* __restrict__ S, float* __restrict__ Dinv) {
  int i = (int)(blockIdx.x * blockDim.x + threadIdx.x);
  if (i >= N_ROWS) return;
  float s = S[i]; if (s == 0.f) s = 1.f;
  Dinv[i] = 1.0f / sqrtf(s);
}

// ---------------------------------------------------------------------------
// 4) Label setup: class counts, Y -> r = p = Y, x = 0.
// ---------------------------------------------------------------------------
__global__ void countsK(const int* __restrict__ labels, const unsigned char* __restrict__ mb,
                        float* __restrict__ counts) {
  int i = (int)(blockIdx.x * blockDim.x + threadIdx.x);
  if (i >= N_ROWS) return;
  if (mb[i]) atomicAdd(&counts[labels[i]], 1.0f);
}

__global__ void initYK(const int* __restrict__ labels, const unsigned char* __restrict__ mb,
                       const float* __restrict__ counts,
                       float* __restrict__ x, float* __restrict__ r, float* __restrict__ p) {
  int g = (int)(blockIdx.x * blockDim.x + threadIdx.x);
  if (g >= N_ROWS * NCLS) return;
  int i = g / NCLS, c = g % NCLS;
  float y = (mb[i] && labels[i] == c) ? (1.0f / fmaxf(counts[c], 1.0f)) : 0.0f;
  x[g] = 0.f; r[g] = y; p[g] = y;
}

// Column-wise dot: out[c] = sum_i A[i][c]*B[i][c].  One block per class.
__global__ __launch_bounds__(256) void dotColsK(const float* __restrict__ A,
                                                const float* __restrict__ B,
                                                float* __restrict__ out) {
  __shared__ float red[256];
  int c = (int)blockIdx.x;
  float acc = 0.f;
  for (int i = (int)threadIdx.x; i < N_ROWS; i += 256)
    acc += A[(size_t)i * NCLS + c] * B[(size_t)i * NCLS + c];
  red[threadIdx.x] = acc; __syncthreads();
  for (int s = 128; s > 0; s >>= 1) {
    if ((int)threadIdx.x < s) red[threadIdx.x] += red[threadIdx.x + s];
    __syncthreads();
  }
  if (threadIdx.x == 0) out[c] = red[0];
}

__global__ void copy100K(const float* __restrict__ src, float* __restrict__ dst) {
  int c = (int)threadIdx.x; if (c < NCLS) dst[c] = src[c];
}

// ---------------------------------------------------------------------------
// 5) CG iteration kernels
// ---------------------------------------------------------------------------
__global__ void uK(const float* __restrict__ Dinv, const float* __restrict__ p,
                   float* __restrict__ U) {
  int g = (int)(blockIdx.x * blockDim.x + threadIdx.x);
  if (g >= N_ROWS * NCLS) return;
  U[g] = Dinv[g / NCLS] * p[g];
}

__global__ void scatterK(const float* __restrict__ U, const float* __restrict__ eW,
                         const int* __restrict__ eC, float* __restrict__ WU) {
  int e = (int)(blockIdx.x * blockDim.x + threadIdx.x);
  if (e >= N_ROWS * KNN) return;
  float w = eW[e];
  if (w == 0.f) return;
  int r = e / KNN, c = eC[e];
  const float* Ur = U + (size_t)r * NCLS;
  const float* Uc = U + (size_t)c * NCLS;
  float* Wr = WU + (size_t)r * NCLS;
  float* Wc = WU + (size_t)c * NCLS;
  for (int k = 0; k < NCLS; ++k) {
    atomicAdd(&Wr[k], w * Uc[k]);
    atomicAdd(&Wc[k], w * Ur[k]);
  }
}

__global__ void apK(const float* __restrict__ p, const float* __restrict__ Dinv,
                    const float* __restrict__ WU, float* __restrict__ Ap) {
  int g = (int)(blockIdx.x * blockDim.x + threadIdx.x);
  if (g >= N_ROWS * NCLS) return;
  Ap[g] = p[g] - ALPHA_C * Dinv[g / NCLS] * WU[g];
}

__global__ void alphaK(const float* __restrict__ rs, const float* __restrict__ bnorm2,
                       const float* __restrict__ pAp, float* __restrict__ a,
                       int* __restrict__ active) {
  int c = (int)threadIdx.x; if (c >= NCLS) return;
  int act = sqrtf(rs[c]) > TOL_C * fmaxf(sqrtf(bnorm2[c]), 1e-30f);
  active[c] = act;
  a[c] = act ? rs[c] / fmaxf(pAp[c], 1e-30f) : 0.0f;
}

__global__ void updXRK(const float* __restrict__ a, const float* __restrict__ p,
                       const float* __restrict__ Ap, float* __restrict__ x,
                       float* __restrict__ r) {
  int g = (int)(blockIdx.x * blockDim.x + threadIdx.x);
  if (g >= N_ROWS * NCLS) return;
  float ac = a[g % NCLS];
  x[g] += ac * p[g];
  r[g] -= ac * Ap[g];
}

__global__ void betaK(const int* __restrict__ active, const float* __restrict__ rsNew,
                      float* __restrict__ rs, float* __restrict__ beta) {
  int c = (int)threadIdx.x; if (c >= NCLS) return;
  if (active[c]) { beta[c] = rsNew[c] / fmaxf(rs[c], 1e-30f); rs[c] = rsNew[c]; }
  else beta[c] = 0.0f;
}

__global__ void updPK(const int* __restrict__ active, const float* __restrict__ beta,
                      const float* __restrict__ r, float* __restrict__ p) {
  int g = (int)(blockIdx.x * blockDim.x + threadIdx.x);
  if (g >= N_ROWS * NCLS) return;
  int c = g % NCLS;
  if (active[c]) p[g] = r[g] + beta[c] * p[g];
}

// ---------------------------------------------------------------------------
// 6) Final: relu -> L1 -> softmax -> argmax/maxprob/mask/acc + label override
// ---------------------------------------------------------------------------
__global__ void finalK(const float* __restrict__ x, const int* __restrict__ labels,
                       const unsigned char* __restrict__ mb,
                       float* __restrict__ outP, float* __restrict__ outM,
                       float* __restrict__ accBuf) {
  int i = (int)(blockIdx.x * blockDim.x + threadIdx.x);
  if (i >= N_ROWS) return;
  const float* xi = x + (size_t)i * NCLS;
  float s1 = 0.f, mz = -1e30f; int am = 0;
  for (int c = 0; c < NCLS; ++c) {
    float z = fmaxf(xi[c], 0.f);
    s1 += z;
    if (z > mz) { mz = z; am = c; }
  }
  float inv = 1.0f / fmaxf(s1, 1e-12f);
  float ml = mz * inv;
  float se = 0.f;
  for (int c = 0; c < NCLS; ++c) se += expf(fmaxf(xi[c], 0.f) * inv - ml);
  float maxp = 1.0f / se;                 // softmax value at the argmax
  float mk = (maxp >= THRESH_C) ? 1.f : 0.f;
  atomicAdd(accBuf, (am == labels[i]) ? 1.0f : 0.0f);
  int pl = mb[i] ? labels[i] : am;
  outP[i] = (float)pl;
  outM[i] = mk;
}

__global__ void accFinalK(const float* __restrict__ accBuf, float* __restrict__ out) {
  out[0] = accBuf[0] * (1.0f / (float)N_ROWS);
}

// ---------------------------------------------------------------------------
// Host launcher
// ---------------------------------------------------------------------------
static inline void* wsAlloc(char*& p, size_t bytes) {
  void* r = (void*)p;
  p += (bytes + 255) & ~((size_t)255);
  return r;
}

extern "C" void kernel_launch(void* const* d_in, const int* in_sizes, int n_in,
                              void* d_out, int out_size, void* d_ws, size_t ws_size,
                              hipStream_t stream) {
  (void)in_sizes; (void)n_in; (void)out_size; (void)ws_size;
  const float*         X      = (const float*)d_in[0];
  const int*           labels = (const int*)d_in[1];
  const unsigned char* maskb  = (const unsigned char*)d_in[2];   // jax bool = 1 byte

  char* wp = (char*)d_ws;
  _Float16* Xn   = (_Float16*)wsAlloc(wp, (size_t)N_ROWS * DIMX * sizeof(_Float16));
  float* topV    = (float*)wsAlloc(wp, (size_t)N_ROWS * KSEL * 4);
  int*   topI    = (int*)  wsAlloc(wp, (size_t)N_ROWS * KSEL * 4);
  float* eW      = (float*)wsAlloc(wp, (size_t)N_ROWS * KNN * 4);
  int*   eC      = (int*)  wsAlloc(wp, (size_t)N_ROWS * KNN * 4);
  float* S       = (float*)wsAlloc(wp, (size_t)N_ROWS * 4);
  float* Dinv    = (float*)wsAlloc(wp, (size_t)N_ROWS * 4);
  float* counts  = (float*)wsAlloc(wp, NCLS * 4);
  float* rs      = (float*)wsAlloc(wp, NCLS * 4);
  float* bnorm2  = (float*)wsAlloc(wp, NCLS * 4);
  float* rsNew   = (float*)wsAlloc(wp, NCLS * 4);
  float* pAp     = (float*)wsAlloc(wp, NCLS * 4);
  float* aC      = (float*)wsAlloc(wp, NCLS * 4);
  float* betaC   = (float*)wsAlloc(wp, NCLS * 4);
  int*   active  = (int*)  wsAlloc(wp, NCLS * 4);
  float* accBuf  = (float*)wsAlloc(wp, 4);
  float* xv      = (float*)wsAlloc(wp, (size_t)N_ROWS * NCLS * 4);
  float* rv      = (float*)wsAlloc(wp, (size_t)N_ROWS * NCLS * 4);
  float* pv      = (float*)wsAlloc(wp, (size_t)N_ROWS * NCLS * 4);
  float* Apv     = (float*)wsAlloc(wp, (size_t)N_ROWS * NCLS * 4);
  float* WU      = (float*)wsAlloc(wp, (size_t)N_ROWS * NCLS * 4);
  float* Uv      = (float*)wsAlloc(wp, (size_t)N_ROWS * NCLS * 4);

  const int NC = N_ROWS * NCLS;           // 1,638,400
  const int ecount = N_ROWS * KNN;        // 819,200

  hipMemsetAsync(S, 0, (size_t)N_ROWS * 4, stream);
  hipMemsetAsync(counts, 0, NCLS * 4, stream);
  hipMemsetAsync(accBuf, 0, 4, stream);

  normalizeK<<<N_ROWS / 8, 256, 0, stream>>>(X, Xn);
  gemm_topk<<<N_ROWS / (16 * GT_WAVES), 128, 0, stream>>>(Xn, topV, topI);
  sortEdgesK<<<(N_ROWS + 127) / 128, 128, 0, stream>>>(topV, topI, eW, eC, S);
  dinvK<<<(N_ROWS + 255) / 256, 256, 0, stream>>>(S, Dinv);
  countsK<<<(N_ROWS + 255) / 256, 256, 0, stream>>>(labels, maskb, counts);
  initYK<<<(NC + 255) / 256, 256, 0, stream>>>(labels, maskb, counts, xv, rv, pv);
  dotColsK<<<NCLS, 256, 0, stream>>>(rv, rv, rs);        // rs0 = |Y|^2 per class
  copy100K<<<1, 128, 0, stream>>>(rs, bnorm2);           // bnorm^2 == rs0

  for (int it = 0; it < 20; ++it) {
    uK<<<(NC + 255) / 256, 256, 0, stream>>>(Dinv, pv, Uv);
    hipMemsetAsync(WU, 0, (size_t)NC * 4, stream);
    scatterK<<<(ecount + 255) / 256, 256, 0, stream>>>(Uv, eW, eC, WU);
    apK<<<(NC + 255) / 256, 256, 0, stream>>>(pv, Dinv, WU, Apv);
    dotColsK<<<NCLS, 256, 0, stream>>>(pv, Apv, pAp);
    alphaK<<<1, 128, 0, stream>>>(rs, bnorm2, pAp, aC, active);
    updXRK<<<(NC + 255) / 256, 256, 0, stream>>>(aC, pv, Apv, xv, rv);
    dotColsK<<<NCLS, 256, 0, stream>>>(rv, rv, rsNew);
    betaK<<<1, 128, 0, stream>>>(active, rsNew, rs, betaC);
    updPK<<<(NC + 255) / 256, 256, 0, stream>>>(active, betaC, rv, pv);
  }

  float* out = (float*)d_out;             // [p_labels(N) | acc | masks(N)]
  finalK<<<(N_ROWS + 127) / 128, 128, 0, stream>>>(xv, labels, maskb,
                                                   out, out + N_ROWS + 1, accBuf);
  accFinalK<<<1, 1, 0, stream>>>(accBuf, out + N_ROWS);
}
